// VQVAE_MLP_20693152432836
// MI455X (gfx1250) — matrix operand order repaired and checked by reading, not compiled, not run on hardware
//
#include <hip/hip_runtime.h>

// ---------------------------------------------------------------------------
// VQ-VAE MLP forward for MI455X (gfx1250, wave32, WMMA bf16 -> f32 accum)
// ---------------------------------------------------------------------------

typedef __bf16 bf16;
typedef __attribute__((ext_vector_type(16))) __bf16 v16bf;
typedef __attribute__((ext_vector_type(8)))  __bf16 v8bf;
typedef __attribute__((ext_vector_type(8)))  float  v8f;

#define BATCH   1024
#define PATCH   512
#define EMB     64
#define NCODE   1024
#define LSEQ    128
#define HID     1024
#define NROWS   (BATCH * LSEQ)      // 131072 VQ rows
#define NELEM   (NROWS * EMB)       // 8388608

#define LDSS    40                  // GEMM LDS row stride in bf16 (32 data + 8 pad)

__device__ __forceinline__ v16bf make_frag(const bf16* p0, const bf16* p1) {
  v8bf lo = *(const v8bf*)p0;
  v8bf hi = *(const v8bf*)p1;
  return __builtin_shufflevector(lo, hi, 0,1,2,3,4,5,6,7,8,9,10,11,12,13,14,15);
}

__device__ __forceinline__ float gelu_f(float x) {
  return 0.5f * x * (1.0f + erff(x * 0.70710678118654752440f));
}

// ---------------------------------------------------------------------------
// WMMA GEMM: C[M,N] = A[M,K] * W[N,K]^T + bias[N]
// fp32 global operands converted to bf16 while staging to LDS.
// Block: 256 threads (8 waves). Tile 128x128, wave tile 32x64, K-step 32.
// ---------------------------------------------------------------------------
__global__ __launch_bounds__(256)
void gemm_wmma(const float* __restrict__ A, const float* __restrict__ W,
               const float* __restrict__ bias, float* __restrict__ C,
               int M, int N, int Kd) {
  __shared__ alignas(16) bf16 At[128 * LDSS];
  __shared__ alignas(16) bf16 Wt[128 * LDSS];

  const int tid  = threadIdx.x;
  const int lane = tid & 31;
  const int wave = tid >> 5;
  const int rowBase = (wave >> 1) * 32;   // 4 waves along M
  const int colBase = (wave & 1) * 64;    // 2 waves along N
  const int gRow0 = blockIdx.y * 128;
  const int gCol0 = blockIdx.x * 128;

  const int mh  = lane & 15;
  const int kbA = (lane >> 4) * 16;       // A frag: K {kbA..kbA+15}
  const int kbB = (lane >> 4) * 16;       // B frag: K {kbB..kbB+15}

  v8f acc[2][4] = {};

  const int nChunks = Kd >> 5;
  for (int kc = 0; kc < nChunks; ++kc) {
    const int k0 = kc << 5;
    // Stage 128x32 fp32 of A and W -> bf16 LDS. 512 8-float chunks each.
#pragma unroll
    for (int it = 0; it < 2; ++it) {
      const int ch  = tid + it * 256;
      const int r   = ch >> 2;
      const int off = (ch & 3) * 8;
      const float* sa = A + (size_t)(gRow0 + r) * Kd + k0 + off;
      float4 a0 = *(const float4*)sa;
      float4 a1 = *(const float4*)(sa + 4);
      v8bf pa = {(bf16)a0.x,(bf16)a0.y,(bf16)a0.z,(bf16)a0.w,
                 (bf16)a1.x,(bf16)a1.y,(bf16)a1.z,(bf16)a1.w};
      *(v8bf*)(At + r * LDSS + off) = pa;
      const float* sw = W + (size_t)(gCol0 + r) * Kd + k0 + off;
      float4 w0 = *(const float4*)sw;
      float4 w1 = *(const float4*)(sw + 4);
      v8bf pw = {(bf16)w0.x,(bf16)w0.y,(bf16)w0.z,(bf16)w0.w,
                 (bf16)w1.x,(bf16)w1.y,(bf16)w1.z,(bf16)w1.w};
      *(v8bf*)(Wt + r * LDSS + off) = pw;
    }
    __syncthreads();

    if (kc + 1 < nChunks) {   // prefetch next K-chunk (global_prefetch_b8)
#pragma unroll
      for (int it = 0; it < 2; ++it) {
        const int ch  = tid + it * 256;
        const int r   = ch >> 2;
        const int off = (ch & 3) * 8;
        __builtin_prefetch(A + (size_t)(gRow0 + r) * Kd + k0 + 32 + off, 0, 0);
        __builtin_prefetch(W + (size_t)(gCol0 + r) * Kd + k0 + 32 + off, 0, 0);
      }
    }

    v16bf af[2], bfr[4];
#pragma unroll
    for (int i = 0; i < 2; ++i) {
      const bf16* ar = At + (rowBase + i * 16 + mh) * LDSS;
      af[i] = make_frag(ar + kbA, ar + kbA + 8);
    }
#pragma unroll
    for (int j = 0; j < 4; ++j) {
      const bf16* br = Wt + (colBase + j * 16 + mh) * LDSS;
      bfr[j] = make_frag(br + kbB, br + kbB + 8);
    }
#pragma unroll
    for (int i = 0; i < 2; ++i)
#pragma unroll
      for (int j = 0; j < 4; ++j)
        acc[i][j] = __builtin_amdgcn_wmma_f32_16x16x32_bf16(
            false, af[i], false, bfr[j], (short)0, acc[i][j], false, false);
    __syncthreads();
  }

  // C/D layout: lanes 0-15 -> M=v, lanes 16-31 -> M=v+8; N = lane&15.
  const int rsel = (lane < 16) ? 0 : 8;
#pragma unroll
  for (int i = 0; i < 2; ++i) {
#pragma unroll
    for (int j = 0; j < 4; ++j) {
      const int col = gCol0 + colBase + j * 16 + mh;
      const float bv = bias[col];
#pragma unroll
      for (int v = 0; v < 8; ++v) {
        const int row = gRow0 + rowBase + i * 16 + rsel + v;
        C[(size_t)row * N + col] = acc[i][j][v] + bv;
      }
    }
  }
}

// ---------------------------------------------------------------------------
// LayerNorm over last dim (=1024) with optional exact-erf GELU.
// ---------------------------------------------------------------------------
__global__ __launch_bounds__(256)
void ln1024(const float* __restrict__ X, const float* __restrict__ g,
            const float* __restrict__ b, float* __restrict__ Y, int doGelu) {
  __shared__ float red[256];
  const int row = blockIdx.x;
  const float* x = X + (size_t)row * HID;
  float v0[4]; float s = 0.f;
#pragma unroll
  for (int i = 0; i < 4; ++i) { v0[i] = x[threadIdx.x + i * 256]; s += v0[i]; }
  red[threadIdx.x] = s; __syncthreads();
  for (int o = 128; o > 0; o >>= 1) {
    if (threadIdx.x < o) red[threadIdx.x] += red[threadIdx.x + o];
    __syncthreads();
  }
  const float mean = red[0] * (1.0f / HID);
  __syncthreads();
  float vs = 0.f;
#pragma unroll
  for (int i = 0; i < 4; ++i) { float d = v0[i] - mean; vs += d * d; }
  red[threadIdx.x] = vs; __syncthreads();
  for (int o = 128; o > 0; o >>= 1) {
    if (threadIdx.x < o) red[threadIdx.x] += red[threadIdx.x + o];
    __syncthreads();
  }
  const float rstd = rsqrtf(red[0] * (1.0f / HID) + 1e-5f);
#pragma unroll
  for (int i = 0; i < 4; ++i) {
    const int c = threadIdx.x + i * 256;
    float y = (v0[i] - mean) * rstd * g[c] + b[c];
    if (doGelu) y = gelu_f(y);
    Y[(size_t)row * HID + c] = y;
  }
}

__global__ __launch_bounds__(256)
void add_gelu(float* __restrict__ h, const float* __restrict__ t, int n) {
  const int i = blockIdx.x * 256 + threadIdx.x;
  if (i < n) h[i] = gelu_f(h[i] + t[i]);
}

// ---------------------------------------------------------------------------
// VQ: per block = one batch element (128 rows). score = ||c||^2 - 2 z.c via
// WMMA bf16; running per-lane argmin matching the D-matrix VGPR layout.
// ---------------------------------------------------------------------------
__global__ __launch_bounds__(256)
void vq_kernel(const float* __restrict__ z, const float* __restrict__ cb,
               const float* __restrict__ cbnorm, int* __restrict__ idxOut) {
  __shared__ alignas(16) bf16 zT[128 * 72];
  __shared__ alignas(16) bf16 cT[128 * 72];
  __shared__ float cn[NCODE];
  __shared__ float sS[8 * 256];
  __shared__ int   sI[8 * 256];

  const int tid = threadIdx.x, lane = tid & 31, wave = tid >> 5;
  const int bb  = blockIdx.x;                    // batch element
  const float* zb = z + (size_t)bb * (EMB * LSEQ);

  // zT[l][e] = z[b][e*128 + l]  (the [B,EMB,L] -> [B,L,EMB] permute)
  for (int i = tid; i < EMB * LSEQ; i += 256) {
    const int e = i >> 7, l = i & 127;
    zT[l * 72 + e] = (bf16)zb[i];
  }
  for (int i = tid; i < NCODE; i += 256) cn[i] = cbnorm[i];
  __syncthreads();

  const int mh  = lane & 15;
  const int kbA = (lane >> 4) * 16;
  const int kbB = (lane >> 4) * 16;
  const bf16* ar = zT + (wave * 16 + mh) * 72;
  const v16bf a0 = make_frag(ar + kbA,      ar + kbA + 8);
  const v16bf a1 = make_frag(ar + 32 + kbA, ar + 32 + kbA + 8);

  float best[8]; int bidx[8];
#pragma unroll
  for (int v = 0; v < 8; ++v) { best[v] = 3.4e38f; bidx[v] = 0; }

  for (int ct = 0; ct < 8; ++ct) {
    const int c0 = ct * 128;
    __syncthreads();
#pragma unroll
    for (int it = 0; it < 4; ++it) {            // stage 128x64 codes -> bf16
      const int base = (tid + it * 256) * 8;
      const int n = base >> 6, e = base & 63;
      const float* src = cb + (size_t)(c0 + n) * EMB + e;
      float4 f0 = *(const float4*)src;
      float4 f1 = *(const float4*)(src + 4);
      v8bf pk = {(bf16)f0.x,(bf16)f0.y,(bf16)f0.z,(bf16)f0.w,
                 (bf16)f1.x,(bf16)f1.y,(bf16)f1.z,(bf16)f1.w};
      *(v8bf*)(cT + n * 72 + e) = pk;
    }
    __syncthreads();
#pragma unroll
    for (int j = 0; j < 8; ++j) {
      const bf16* br = cT + (j * 16 + mh) * 72;
      v16bf b0 = make_frag(br + kbB,      br + kbB + 8);
      v16bf b1 = make_frag(br + 32 + kbB, br + 32 + kbB + 8);
      v8f acc = {};
      acc = __builtin_amdgcn_wmma_f32_16x16x32_bf16(false, a0, false, b0,
                                                    (short)0, acc, false, false);
      acc = __builtin_amdgcn_wmma_f32_16x16x32_bf16(false, a1, false, b1,
                                                    (short)0, acc, false, false);
      const int col = c0 + j * 16 + mh;
      const float cnv = cn[col];
#pragma unroll
      for (int v = 0; v < 8; ++v) {
        const float s = cnv - 2.0f * acc[v];
        if (s < best[v]) { best[v] = s; bidx[v] = col; }
      }
    }
  }

  const int rsel = (lane < 16) ? 0 : 8;
#pragma unroll
  for (int v = 0; v < 8; ++v) {
    const int m = rsel + v;
    sS[wave * 256 + m * 16 + mh] = best[v];
    sI[wave * 256 + m * 16 + mh] = bidx[v];
  }
  __syncthreads();
  if (lane < 16) {
    float bs = 3.4e38f; int bi = 0;
    const float* ss = sS + wave * 256 + lane * 16;
    const int*   si = sI + wave * 256 + lane * 16;
#pragma unroll
    for (int n = 0; n < 16; ++n)
      if (ss[n] < bs) { bs = ss[n]; bi = si[n]; }
    idxOut[bb * LSEQ + wave * 16 + lane] = bi;
  }
}

// ---------------------------------------------------------------------------
// Small helper kernels
// ---------------------------------------------------------------------------
__global__ void zero_misc(int* __restrict__ hist, float* __restrict__ accum) {
  const int i = blockIdx.x * 256 + threadIdx.x;
  if (i < NCODE) hist[i] = 0;
  if (i == 0) accum[0] = 0.f;
}

__global__ void cbnorm_kernel(const float* __restrict__ cb, float* __restrict__ cbn) {
  const int k = blockIdx.x * 256 + threadIdx.x;
  if (k < NCODE) {
    float s = 0.f;
    const float* c = cb + (size_t)k * EMB;
#pragma unroll
    for (int e = 0; e < EMB; ++e) s += c[e] * c[e];
    cbn[k] = s;
  }
}

// loss accumulate: sum of (codebook[idx[r]][e] - z[b][e*128+l])^2
__global__ __launch_bounds__(256)
void loss_kernel(const float* __restrict__ z, const float* __restrict__ cb,
                 const int* __restrict__ idx, float* __restrict__ accum) {
  __shared__ float red[256];
  const int g0 = blockIdx.x * 256 * 8;
  float s = 0.f;
#pragma unroll
  for (int it = 0; it < 8; ++it) {
    const int g = g0 + it * 256 + threadIdx.x;
    const int r = g >> 6, e = g & 63;
    const int b = r >> 7, l = r & 127;
    const float q  = cb[(size_t)idx[r] * EMB + e];
    const float zz = z[(size_t)b * (EMB * LSEQ) + e * LSEQ + l];
    const float d = q - zz;
    s += d * d;
  }
  red[threadIdx.x] = s; __syncthreads();
  for (int o = 128; o > 0; o >>= 1) {
    if (threadIdx.x < o) red[threadIdx.x] += red[threadIdx.x + o];
    __syncthreads();
  }
  if (threadIdx.x == 0) atomicAdd(accum, red[0]);
}

__global__ void finalize_loss(const float* __restrict__ accum, float* __restrict__ out0) {
  // loss = q_latent + 0.25*e_latent ; both equal MSE in forward value
  out0[0] = 1.25f * accum[0] * (1.0f / (float)NELEM);
}

__global__ void hist_kernel(const int* __restrict__ idx, int* __restrict__ hist) {
  const int i = blockIdx.x * 256 + threadIdx.x;
  if (i < NROWS) atomicAdd(&hist[idx[i]], 1);
}

__global__ __launch_bounds__(256)
void perp_kernel(const int* __restrict__ hist, float* __restrict__ outp) {
  __shared__ float red[256];
  float s = 0.f;
  for (int i = threadIdx.x; i < NCODE; i += 256) {
    const float p = (float)hist[i] * (1.0f / (float)NROWS);
    s += -p * logf(p + 1e-10f);
  }
  red[threadIdx.x] = s; __syncthreads();
  for (int o = 128; o > 0; o >>= 1) {
    if (threadIdx.x < o) red[threadIdx.x] += red[threadIdx.x + o];
    __syncthreads();
  }
  if (threadIdx.x == 0) outp[0] = expf(red[0]);
}

// dec_in[b][e*128 + l] = codebook[idx[b*128+l]][e]   (straight-through == quantized)
__global__ __launch_bounds__(256)
void gather_kernel(const float* __restrict__ cb, const int* __restrict__ idx,
                   float* __restrict__ din) {
  const int g = blockIdx.x * 256 + threadIdx.x;
  const int b = g >> 13, o = g & 8191;
  const int e = o >> 7, l = o & 127;
  din[g] = cb[(size_t)idx[b * LSEQ + l] * EMB + e];
}

// ---------------------------------------------------------------------------
// Host side
// ---------------------------------------------------------------------------
static inline const float* Pp(void* const* d_in, int i) { return (const float*)d_in[i]; }

static void run_mlp(void* const* d_in, int base, const float* X, int Kin,
                    float* h, float* t, float* c, float* out, int Nout,
                    hipStream_t s) {
  const int M = BATCH;
  dim3 blk(256);
  gemm_wmma<<<dim3(HID / 128, M / 128), blk, 0, s>>>(X, Pp(d_in, base + 0), Pp(d_in, base + 1), c, M, HID, Kin);
  ln1024<<<M, blk, 0, s>>>(c, Pp(d_in, base + 2), Pp(d_in, base + 3), h, 1);
  for (int i = 0; i < 3; ++i) {
    const int bb = base + 4 + 8 * i;
    gemm_wmma<<<dim3(HID / 128, M / 128), blk, 0, s>>>(h, Pp(d_in, bb + 0), Pp(d_in, bb + 1), c, M, HID, HID);
    ln1024<<<M, blk, 0, s>>>(c, Pp(d_in, bb + 2), Pp(d_in, bb + 3), t, 1);
    gemm_wmma<<<dim3(HID / 128, M / 128), blk, 0, s>>>(t, Pp(d_in, bb + 4), Pp(d_in, bb + 5), c, M, HID, HID);
    ln1024<<<M, blk, 0, s>>>(c, Pp(d_in, bb + 6), Pp(d_in, bb + 7), t, 0);
    add_gelu<<<(M * HID) / 256, blk, 0, s>>>(h, t, M * HID);
  }
  const int ob = base + 28;
  gemm_wmma<<<dim3(HID / 128, M / 128), blk, 0, s>>>(h, Pp(d_in, ob + 0), Pp(d_in, ob + 1), c, M, HID, HID);
  ln1024<<<M, blk, 0, s>>>(c, Pp(d_in, ob + 2), Pp(d_in, ob + 3), t, 1);
  gemm_wmma<<<dim3(Nout / 128, M / 128), blk, 0, s>>>(t, Pp(d_in, ob + 4), Pp(d_in, ob + 5), out, M, Nout, HID);
}

extern "C" void kernel_launch(void* const* d_in, const int* in_sizes, int n_in,
                              void* d_out, int out_size, void* d_ws, size_t ws_size,
                              hipStream_t stream) {
  (void)in_sizes; (void)n_in; (void)out_size; (void)ws_size;
  const float* x  = (const float*)d_in[0];
  const float* cb = (const float*)d_in[35];
  float* out = (float*)d_out;            // [loss | x_recon(1024x512) | perplexity]
  float* ws  = (float*)d_ws;

  const size_t M1 = (size_t)1 << 20;     // 1M floats
  float* h   = ws;                       // [1M]  residual stream
  float* t   = ws + M1;                  // [1M]  block temp
  float* c   = ws + 2 * M1;              // [1M]  GEMM out (pre-LN)
  float* z   = ws + 3 * M1;              // [8M]  encoder out; reused as dec input
  float* cbn = ws + 11 * M1;             // [1024]
  float* acc = ws + 11 * M1 + 1024;      // [1]
  int* hist  = (int*)(ws + 11 * M1 + 2048);   // [1024]
  int* idx   = (int*)(ws + 11 * M1 + 3072);   // [131072]

  dim3 blk(256);
  zero_misc<<<4, blk, 0, stream>>>(hist, acc);
  cbnorm_kernel<<<4, blk, 0, stream>>>(cb, cbn);

  // encoder: x[1024,512] -> z[1024, 8192]
  run_mlp(d_in, 1, x, PATCH, h, t, c, z, EMB * LSEQ, stream);

  // VQ
  vq_kernel<<<BATCH, blk, 0, stream>>>(z, cb, cbn, idx);
  loss_kernel<<<NELEM / (256 * 8), blk, 0, stream>>>(z, cb, idx, acc);
  finalize_loss<<<1, 1, 0, stream>>>(acc, out);
  hist_kernel<<<NROWS / 256, blk, 0, stream>>>(idx, hist);
  perp_kernel<<<1, blk, 0, stream>>>(hist, out + 1 + BATCH * PATCH);

  // build decoder input (== quantized, straight-through) into z's storage
  gather_kernel<<<NELEM / 256, blk, 0, stream>>>(cb, idx, z);

  // decoder: z[1024, 8192] -> x_recon[1024,512] at out+1
  run_mlp(d_in, 36, z, EMB * LSEQ, h, t, c, out + 1, PATCH, stream);
}
